// MMGCN_36249523978808
// MI455X (gfx1250) — matrix operand order repaired and checked by reading, not compiled, not run on hardware
//
#include <hip/hip_runtime.h>
#include <hip/hip_bf16.h>

// ---------------------------------------------------------------------------
// MMGCN forward for MI455X (gfx1250, wave32).
// GEMMs run on v_wmma_f32_16x16x32_bf16 (bf16 in, fp32 accumulate).
// Edge aggregation uses native global_atomic_add_f32 into an L2-resident
// accumulator (per-pass working set ~100MB < 192MB L2).
// ---------------------------------------------------------------------------

typedef __attribute__((ext_vector_type(16))) __bf16 v16bf;
typedef __attribute__((ext_vector_type(8)))  float  v8f;
typedef unsigned short u16;

union FragBF { v16bf v; uint4 q[2]; };

__device__ __forceinline__ float lrelu(float v) { return v > 0.0f ? v : 0.01f * v; }

__device__ __forceinline__ u16 f2bf(float f) {
    unsigned int u = __float_as_uint(f);
    u += 0x7fffu + ((u >> 16) & 1u);            // round to nearest even
    return (u16)(u >> 16);
}

// ---------------------------------------------------------------------------
__global__ void k_zero_u32(unsigned* __restrict__ p, int n) {
    int i = blockIdx.x * blockDim.x + threadIdx.x;
    if (i < n) p[i] = 0u;
}

__global__ void k_zero_f4(float4* __restrict__ p, int n4) {
    int i = blockIdx.x * blockDim.x + threadIdx.x;
    if (i < n4) p[i] = make_float4(0.f, 0.f, 0.f, 0.f);
}

__global__ void k_degree(const int* __restrict__ src, const int* __restrict__ dst,
                         unsigned* __restrict__ deg, int E, int N) {
    int e = blockIdx.x * blockDim.x + threadIdx.x;
    if (e >= E) return;
    atomicAdd(&deg[src[e]], 1u);
    atomicAdd(&deg[N + dst[e]], 1u);
}

__global__ void k_norms(const unsigned* __restrict__ deg,
                        float* __restrict__ onorm, float* __restrict__ inorm, int N) {
    int i = blockIdx.x * blockDim.x + threadIdx.x;
    if (i >= N) return;
    onorm[i] = rsqrtf(fmaxf((float)deg[i], 1.0f));
    inorm[i] = rsqrtf(fmaxf((float)deg[N + i], 1.0f));
}

__global__ void k_f2bf(const float* __restrict__ s, u16* __restrict__ d, int n) {
    int i = blockIdx.x * blockDim.x + threadIdx.x;
    if (i < n) d[i] = f2bf(s[i]);
}

// ---------------------------------------------------------------------------
// Row L2-normalize. One wave per row (32 lanes x float4 = 128 cols).
// Emits: out[:, :128] = leaky(xn)   (identical in both branches -> exact)
//        A0  = bf16(xn * out_norm)            (layer-0 GEMM operand)
//        A1c = bf16(leaky(xn) * out_norm)     (layer-1 operand, K rows 0..127)
__global__ void __launch_bounds__(256)
k_rownorm(const float* __restrict__ x, const float* __restrict__ onorm,
          float* __restrict__ out, u16* __restrict__ A0, u16* __restrict__ A1c, int N) {
    int row = blockIdx.x * 8 + (threadIdx.x >> 5);
    if (row >= N) return;
    int lane = threadIdx.x & 31;
    int c = lane * 4;
    float4 v = *(const float4*)(x + (size_t)row * 128 + c);
    float ss = v.x * v.x + v.y * v.y + v.z * v.z + v.w * v.w;
    #pragma unroll
    for (int o = 16; o > 0; o >>= 1) ss += __shfl_xor(ss, o, 32);
    float s = 1.0f / fmaxf(sqrtf(ss), 1e-12f);
    float a0 = v.x * s, a1 = v.y * s, a2 = v.z * s, a3 = v.w * s;
    float l0 = lrelu(a0), l1 = lrelu(a1), l2 = lrelu(a2), l3 = lrelu(a3);
    // out row stride = 384
    *(float4*)(out + (size_t)row * 384 + c) = make_float4(l0, l1, l2, l3);
    float on = onorm[row];
    ushort4 p0; p0.x = f2bf(a0 * on); p0.y = f2bf(a1 * on); p0.z = f2bf(a2 * on); p0.w = f2bf(a3 * on);
    *(ushort4*)(A0  + (size_t)row * 128 + c) = p0;
    ushort4 p1; p1.x = f2bf(l0 * on); p1.y = f2bf(l1 * on); p1.z = f2bf(l2 * on); p1.w = f2bf(l3 * on);
    *(ushort4*)(A1c + (size_t)row * 128 + c) = p1;
}

// ---------------------------------------------------------------------------
// GEMM: C[M,128] = [A0 | A1] (bf16, row-major, 128 cols each) @ W (bf16, Kx128).
// One wave -> one 16x16 f32 tile via v_wmma_f32_16x16x32_bf16.
// Block = 256 threads = 8 waves = the 8 N-tiles of one 16-row M-tile.
// Fragment layouts per CDNA5 ISA 7.12.2 (16-bit A 16x32, B 32x16, 32-bit C 16x16).
__global__ void __launch_bounds__(256)
k_gemm_bf16(const u16* __restrict__ A0, const u16* __restrict__ A1,
            const u16* __restrict__ W, float* __restrict__ C) {
    const int lane  = threadIdx.x & 31;
    const int wv    = threadIdx.x >> 5;          // N-tile 0..7
    const int mtile = blockIdx.x;
    const int half  = lane >> 4;                 // lane group (0: lanes 0-15, 1: 16-31)
    const size_t arow = (size_t)(mtile * 16 + (lane & 15)) * 128;

    v8f acc = {};
    FragBF a, b;
    const u16* Ap = A0 + arow;
    #pragma unroll
    for (int kb = 0; kb < 128; kb += 32) {
        // A: lane holds row m; elems 0..7 = K kb+8*half.., elems 8..15 = K kb+16+8*half..
        a.q[0] = *(const uint4*)(Ap + kb + 8 * half);
        a.q[1] = *(const uint4*)(Ap + kb + 16 + 8 * half);
        // B: lane holds column N = lane&15; elems 0..7 = K kb+8*half.., elems 8..15 = K kb+16+8*half..
        const u16* Wp = W + (size_t)(kb + 8 * half) * 128 + wv * 16 + (lane & 15);
        b.q[0] = make_uint4(Wp[0] | ((unsigned)Wp[128] << 16), Wp[256] | ((unsigned)Wp[384] << 16), Wp[512] | ((unsigned)Wp[640] << 16), Wp[768] | ((unsigned)Wp[896] << 16));
        b.q[1] = make_uint4(Wp[2048] | ((unsigned)Wp[2176] << 16), Wp[2304] | ((unsigned)Wp[2432] << 16), Wp[2560] | ((unsigned)Wp[2688] << 16), Wp[2816] | ((unsigned)Wp[2944] << 16));
        acc = __builtin_amdgcn_wmma_f32_16x16x32_bf16(false, a.v, false, b.v,
                                                      (short)0, acc, false, false);
    }
    if (A1) {   // layer-1: second half of K (rows 128..255 of W)
        const u16* Ap1 = A1 + arow;
        #pragma unroll
        for (int kb = 0; kb < 128; kb += 32) {
            a.q[0] = *(const uint4*)(Ap1 + kb + 8 * half);
            a.q[1] = *(const uint4*)(Ap1 + kb + 16 + 8 * half);
            const u16* Wp = W + (size_t)(128 + kb + 8 * half) * 128 + wv * 16 + (lane & 15);
            b.q[0] = make_uint4(Wp[0] | ((unsigned)Wp[128] << 16), Wp[256] | ((unsigned)Wp[384] << 16), Wp[512] | ((unsigned)Wp[640] << 16), Wp[768] | ((unsigned)Wp[896] << 16));
            b.q[1] = make_uint4(Wp[2048] | ((unsigned)Wp[2176] << 16), Wp[2304] | ((unsigned)Wp[2432] << 16), Wp[2560] | ((unsigned)Wp[2688] << 16), Wp[2816] | ((unsigned)Wp[2944] << 16));
            acc = __builtin_amdgcn_wmma_f32_16x16x32_bf16(false, a.v, false, b.v,
                                                          (short)0, acc, false, false);
        }
    }
    // C: VGPR r -> M = r + 8*half, N = lane&15
    float* Cp = C + (size_t)(mtile * 16) * 128 + wv * 16 + (lane & 15);
    #pragma unroll
    for (int r = 0; r < 8; ++r)
        Cp[(size_t)(r + 8 * half) * 128] = acc[r];
}

// ---------------------------------------------------------------------------
// Edge scatter-add: agg[dst[e], :] += hW[src[e], :].
// One wave per edge; lane handles 4 contiguous floats (float4 load + 4 fp32
// atomics). src/dst are wave-uniform -> scalarized loads.
__global__ void __launch_bounds__(256)
k_edge(const int* __restrict__ src, const int* __restrict__ dst,
       const float* __restrict__ hW, float* __restrict__ agg, int E) {
    int t = blockIdx.x * blockDim.x + threadIdx.x;
    int e = t >> 5;
    if (e >= E) return;
    int lane = t & 31;
    const float4 v = *(const float4*)(hW + (size_t)src[e] * 128 + lane * 4);
    float* p = agg + (size_t)dst[e] * 128 + lane * 4;
    unsafeAtomicAdd(p + 0, v.x);
    unsafeAtomicAdd(p + 1, v.y);
    unsafeAtomicAdd(p + 2, v.z);
    unsafeAtomicAdd(p + 3, v.w);
}

// ---------------------------------------------------------------------------
// Layer-0 epilogue: xhat1 = agg*in_norm + b0 + id_emb[dst_ids];
// h2 = leaky(xhat1). out[:,128:256] (store if mode==0, += if mode==1) gets
// 0.5*h2; A1s = bf16(h2 * out_norm) feeds the layer-1 GEMM.
__global__ void __launch_bounds__(256)
k_post0(const float* __restrict__ agg, const float* __restrict__ inorm,
        const float* __restrict__ onorm, const float* __restrict__ bias,
        const float* __restrict__ id_emb, const int* __restrict__ dst_ids,
        float* __restrict__ out, u16* __restrict__ A1s, int N, int mode) {
    int t = blockIdx.x * blockDim.x + threadIdx.x;
    int row = t >> 5;
    if (row >= N) return;
    int c = (t & 31) * 4;
    float inn = inorm[row];
    int gd = dst_ids[row];
    float4 g = *(const float4*)(id_emb + (size_t)gd * 128 + c);
    float4 a = *(const float4*)(agg + (size_t)row * 128 + c);
    float4 bb = *(const float4*)(bias + c);
    float h0 = lrelu(a.x * inn + bb.x + g.x);
    float h1 = lrelu(a.y * inn + bb.y + g.y);
    float h2 = lrelu(a.z * inn + bb.z + g.z);
    float h3 = lrelu(a.w * inn + bb.w + g.w);
    float* o = out + (size_t)row * 384 + 128 + c;
    if (mode == 0) {
        *(float4*)o = make_float4(0.5f * h0, 0.5f * h1, 0.5f * h2, 0.5f * h3);
    } else {
        float4 cur = *(float4*)o;
        *(float4*)o = make_float4(cur.x + 0.5f * h0, cur.y + 0.5f * h1,
                                  cur.z + 0.5f * h2, cur.w + 0.5f * h3);
    }
    float on = onorm[row];
    ushort4 pk; pk.x = f2bf(h0 * on); pk.y = f2bf(h1 * on);
    pk.z = f2bf(h2 * on); pk.w = f2bf(h3 * on);
    *(ushort4*)(A1s + (size_t)row * 128 + c) = pk;
}

// Layer-1 epilogue: xhat2 = agg*in_norm + b1 + gid (no activation);
// out[:,256:384] gets 0.5*xhat2 (store or accumulate).
__global__ void __launch_bounds__(256)
k_post1(const float* __restrict__ agg, const float* __restrict__ inorm,
        const float* __restrict__ bias, const float* __restrict__ id_emb,
        const int* __restrict__ dst_ids, float* __restrict__ out, int N, int mode) {
    int t = blockIdx.x * blockDim.x + threadIdx.x;
    int row = t >> 5;
    if (row >= N) return;
    int c = (t & 31) * 4;
    float inn = inorm[row];
    int gd = dst_ids[row];
    float4 g = *(const float4*)(id_emb + (size_t)gd * 128 + c);
    float4 a = *(const float4*)(agg + (size_t)row * 128 + c);
    float4 bb = *(const float4*)(bias + c);
    float h0 = a.x * inn + bb.x + g.x;
    float h1 = a.y * inn + bb.y + g.y;
    float h2 = a.z * inn + bb.z + g.z;
    float h3 = a.w * inn + bb.w + g.w;
    float* o = out + (size_t)row * 384 + 256 + c;
    if (mode == 0) {
        *(float4*)o = make_float4(0.5f * h0, 0.5f * h1, 0.5f * h2, 0.5f * h3);
    } else {
        float4 cur = *(float4*)o;
        *(float4*)o = make_float4(cur.x + 0.5f * h0, cur.y + 0.5f * h1,
                                  cur.z + 0.5f * h2, cur.w + 0.5f * h3);
    }
}

// ---------------------------------------------------------------------------
extern "C" void kernel_launch(void* const* d_in, const int* in_sizes, int n_in,
                              void* d_out, int out_size, void* d_ws, size_t ws_size,
                              hipStream_t stream) {
    const int N = 100000, E = 640000;
    const float* x      = (const float*)d_in[0];
    const float* id_emb = (const float*)d_in[1];
    const float* Wv0 = (const float*)d_in[2];  const float* bv0 = (const float*)d_in[3];
    const float* Wv1 = (const float*)d_in[4];  const float* bv1 = (const float*)d_in[5];
    const float* Wt0 = (const float*)d_in[6];  const float* bt0 = (const float*)d_in[7];
    const float* Wt1 = (const float*)d_in[8];  const float* bt1 = (const float*)d_in[9];
    const int* src     = (const int*)d_in[10];
    const int* dst     = (const int*)d_in[11];
    const int* dst_ids = (const int*)d_in[12];
    float* out = (float*)d_out;

    // workspace carve-up (256B aligned)
    char* ws = (char*)d_ws; size_t off = 0;
    auto take = [&](size_t b) -> char* {
        char* p = ws + off; off = (off + b + 255) & ~(size_t)255; return p;
    };
    unsigned* deg   = (unsigned*)take((size_t)2 * N * 4);
    float*    onorm = (float*)   take((size_t)N * 4);
    float*    inorm = (float*)   take((size_t)N * 4);
    u16*      A0    = (u16*)     take((size_t)N * 128 * 2);
    u16*      A1c   = (u16*)     take((size_t)N * 128 * 2);
    u16*      A1s   = (u16*)     take((size_t)N * 128 * 2);
    float*    hW    = (float*)   take((size_t)N * 128 * 4);
    float*    agg   = (float*)   take((size_t)N * 128 * 4);
    u16*      Wb0v  = (u16*)     take(128 * 128 * 2);
    u16*      Wb0t  = (u16*)     take(128 * 128 * 2);
    u16*      Wb1v  = (u16*)     take(256 * 128 * 2);
    u16*      Wb1t  = (u16*)     take(256 * 128 * 2);

    const int mtiles   = N / 16;        // 6250 (divides exactly)
    const int rowBlks  = (N + 7) / 8;   // wave-per-row kernels
    const int elemBlks = (N * 32 + 255) / 256;  // thread-per-float4 over [N,128]
    const int aggN4    = N * 32;        // float4 count of agg

    // ---- preprocessing -----------------------------------------------------
    k_zero_u32<<<(2 * N + 255) / 256, 256, 0, stream>>>(deg, 2 * N);
    k_degree  <<<(E + 255) / 256, 256, 0, stream>>>(src, dst, deg, E, N);
    k_norms   <<<(N + 255) / 256, 256, 0, stream>>>(deg, onorm, inorm, N);
    k_f2bf<<<(128 * 128 + 255) / 256, 256, 0, stream>>>(Wv0, Wb0v, 128 * 128);
    k_f2bf<<<(128 * 128 + 255) / 256, 256, 0, stream>>>(Wt0, Wb0t, 128 * 128);
    k_f2bf<<<(256 * 128 + 255) / 256, 256, 0, stream>>>(Wv1, Wb1v, 256 * 128);
    k_f2bf<<<(256 * 128 + 255) / 256, 256, 0, stream>>>(Wt1, Wb1t, 256 * 128);
    k_rownorm<<<rowBlks, 256, 0, stream>>>(x, onorm, out, A0, A1c, N);

    // ---- branch v (mode 0: store) ------------------------------------------
    k_gemm_bf16<<<mtiles, 256, 0, stream>>>(A0, nullptr, Wb0v, hW);
    k_zero_f4<<<(aggN4 + 255) / 256, 256, 0, stream>>>((float4*)agg, aggN4);
    k_edge<<<(E * 32) / 256, 256, 0, stream>>>(src, dst, hW, agg, E);
    k_post0<<<elemBlks, 256, 0, stream>>>(agg, inorm, onorm, bv0, id_emb, dst_ids,
                                          out, A1s, N, 0);
    k_gemm_bf16<<<mtiles, 256, 0, stream>>>(A1c, A1s, Wb1v, hW);
    k_zero_f4<<<(aggN4 + 255) / 256, 256, 0, stream>>>((float4*)agg, aggN4);
    k_edge<<<(E * 32) / 256, 256, 0, stream>>>(src, dst, hW, agg, E);
    k_post1<<<elemBlks, 256, 0, stream>>>(agg, inorm, bv1, id_emb, dst_ids, out, N, 0);

    // ---- branch t (mode 1: accumulate) -------------------------------------
    k_gemm_bf16<<<mtiles, 256, 0, stream>>>(A0, nullptr, Wb0t, hW);
    k_zero_f4<<<(aggN4 + 255) / 256, 256, 0, stream>>>((float4*)agg, aggN4);
    k_edge<<<(E * 32) / 256, 256, 0, stream>>>(src, dst, hW, agg, E);
    k_post0<<<elemBlks, 256, 0, stream>>>(agg, inorm, onorm, bt0, id_emb, dst_ids,
                                          out, A1s, N, 1);
    k_gemm_bf16<<<mtiles, 256, 0, stream>>>(A1c, A1s, Wb1t, hW);
    k_zero_f4<<<(aggN4 + 255) / 256, 256, 0, stream>>>((float4*)agg, aggN4);
    k_edge<<<(E * 32) / 256, 256, 0, stream>>>(src, dst, hW, agg, E);
    k_post1<<<elemBlks, 256, 0, stream>>>(agg, inorm, bt1, id_emb, dst_ids, out, N, 1);

    (void)in_sizes; (void)n_in; (void)out_size; (void)ws_size;
}
